// EPMoE_63144609186376
// MI455X (gfx1250) — compile-verified, matrix operand order checked by reading, not run on hardware
//
#include <hip/hip_runtime.h>
#include <hip/hip_bf16.h>
#include <cstdint>

// Problem constants (match reference)
#define T_TOK 4096
#define H_DIM 4096
#define I_DIM 2048
#define E_NUM 8
#define K_TOP 2

#define TILE_M 64
#define TILE_N 64
#define TILE_K 32
#define MAX_TILES ((T_TOK * K_TOP) / TILE_M + E_NUM)   // 136 (worst-case padded M tiles)
#define MAX_PAIRS (MAX_TILES * TILE_M)                 // 8704

typedef __bf16 v16bf __attribute__((ext_vector_type(16)));
typedef __bf16 v8bf  __attribute__((ext_vector_type(8)));
typedef float  v8f   __attribute__((ext_vector_type(8)));
typedef float  v4f   __attribute__((ext_vector_type(4)));
typedef int    v4i   __attribute__((ext_vector_type(4)));

#if __has_builtin(__builtin_amdgcn_global_load_async_to_lds_b128)
#define HAVE_ASYNC_LDS 1
typedef __attribute__((address_space(1))) v4i gas_v4i;   // global (AS1) int4*
typedef __attribute__((address_space(3))) v4i las_v4i;   // LDS (AS3) int4*
#else
#define HAVE_ASYNC_LDS 0
#endif

__device__ __forceinline__ void wait_asynccnt0() {
#if __has_builtin(__builtin_amdgcn_s_wait_asynccnt)
  __builtin_amdgcn_s_wait_asynccnt(0);
#else
  asm volatile("s_wait_asynccnt 0x0" ::: "memory");
#endif
}

// ---------------------------------------------------------------------------
// helpers
// ---------------------------------------------------------------------------
// load 8 fp32, vector-fptrunc to 8 bf16 -> lowers to 4x v_cvt_pk_bf16_f32
__device__ __forceinline__ uint4 ld8_cvt(const float* __restrict__ p) {
  v4f a = *(const v4f*)p;
  v4f b = *(const v4f*)(p + 4);
  v8f f = __builtin_shufflevector(a, b, 0, 1, 2, 3, 4, 5, 6, 7);
  v8bf o = __builtin_convertvector(f, v8bf);
  return __builtin_bit_cast(uint4, o);
}

struct BFrag { uint4 a; uint4 b; };   // 32 bytes == v16bf

// Assemble one 16x32 bf16 fragment row (A: row = lane&15; B: col = lane&15).
// K halves: lanes 0-15 take K {0..7, 16..23}; lanes 16-31 take K {8..15, 24..31}
// => two 16B chunks at dword offsets hi4 and 8+hi4 within the padded LDS row.
__device__ __forceinline__ v16bf ld_frag(const unsigned* __restrict__ rowp, int hi4) {
  BFrag t;
  t.a = *(const uint4*)(rowp + hi4);
  t.b = *(const uint4*)(rowp + 8 + hi4);
  return __builtin_bit_cast(v16bf, t);
}

// ---------------------------------------------------------------------------
// routing / schedule kernels
// ---------------------------------------------------------------------------
__global__ void count_kernel(const int* __restrict__ ids, int* __restrict__ counts) {
  int i = blockIdx.x * blockDim.x + threadIdx.x;
  if (i < T_TOK * K_TOP) atomicAdd(&counts[ids[i]], 1);
}

__global__ void sched_kernel(const int* __restrict__ counts, int* __restrict__ tileoff) {
  if (threadIdx.x == 0) {
    int acc = 0;
    for (int e = 0; e < E_NUM; ++e) {
      tileoff[e] = acc;
      acc += (counts[e] + TILE_M - 1) >> 6;
    }
    tileoff[E_NUM] = acc;
  }
}

__global__ void fill_kernel(const int* __restrict__ ids, const float* __restrict__ tw,
                            const int* __restrict__ tileoff, int* __restrict__ cursor,
                            int* __restrict__ ptok, float* __restrict__ pw) {
  int i = blockIdx.x * blockDim.x + threadIdx.x;
  if (i < T_TOK * K_TOP) {
    int e = ids[i];
    int j = atomicAdd(&cursor[e], 1);
    int slot = tileoff[e] * TILE_M + j;
    ptok[slot] = i >> 1;        // token index (K_TOP == 2)
    pw[slot]   = tw[i];
  }
}

// ---------------------------------------------------------------------------
// G1: h[p, i] = silu(x_t . wi0[e,i,:]) * (x_t . wi1[e,i,:]) * w_p   (bf16 out)
//   block = (64 pairs) x (64 intermediate cols); K loop over H in steps of 32
// ---------------------------------------------------------------------------
__global__ __launch_bounds__(256)
void g1_kernel(const float* __restrict__ x,
               const float* __restrict__ wi0,
               const float* __restrict__ wi1,
               const int* __restrict__ tileoff,
               const int* __restrict__ ptok,
               const float* __restrict__ pw,
               unsigned short* __restrict__ hb) {
  __shared__ __align__(16) unsigned As [2][TILE_M][20];  // 32 bf16 + 8 pad per row
  __shared__ __align__(16) unsigned B0s[2][TILE_N][20];
  __shared__ __align__(16) unsigned B1s[2][TILE_N][20];
  __shared__ int   stok[TILE_M];
  __shared__ float sw  [TILE_M];

  const int mt    = blockIdx.y;
  const int total = tileoff[E_NUM];
  if (mt >= total) return;
  int e = 0;
#pragma unroll
  for (int i = 1; i < E_NUM; ++i) e += (mt >= tileoff[i]) ? 1 : 0;

  const int tid      = threadIdx.x;
  const int pairbase = mt * TILE_M;
  if (tid < TILE_M) {
    int tok   = ptok[pairbase + tid];
    stok[tid] = tok;
    sw[tid]   = (tok < 0) ? 0.0f : pw[pairbase + tid];
  }
  __syncthreads();

  const int n0 = blockIdx.x * TILE_N;
  const float* bp0 = wi0 + ((size_t)e * I_DIM + n0) * H_DIM;
  const float* bp1 = wi1 + ((size_t)e * I_DIM + n0) * H_DIM;

  // loader mapping: 256 threads -> 64 rows x 4 chunks of 8 elements
  const int lrow = tid >> 2;
  const int lq   = tid & 3;
  const int ltok = stok[lrow];
  const float* xap = (ltok >= 0) ? (x + (size_t)ltok * H_DIM + lq * 8) : nullptr;
  const float* b0p = bp0 + (size_t)lrow * H_DIM + lq * 8;
  const float* b1p = bp1 + (size_t)lrow * H_DIM + lq * 8;

  uint4 ra, rb0, rb1;
  auto load_stage = [&](int kt) {
    const int k0 = kt * TILE_K;
    ra = xap ? ld8_cvt(xap + k0) : make_uint4(0u, 0u, 0u, 0u);
    rb0 = ld8_cvt(b0p + k0);
    rb1 = ld8_cvt(b1p + k0);
    __builtin_prefetch((const void*)(b0p + k0 + 2 * TILE_K), 0, 3);
    __builtin_prefetch((const void*)(b1p + k0 + 2 * TILE_K), 0, 3);
  };
  auto store_stage = [&](int buf) {
    *(uint4*)&As [buf][lrow][lq * 4] = ra;
    *(uint4*)&B0s[buf][lrow][lq * 4] = rb0;
    *(uint4*)&B1s[buf][lrow][lq * 4] = rb1;
  };

  // wave mapping: 8 waves -> 4 M-tiles x 2 N-tile-pairs
  const int lane  = tid & 31;
  const int wv    = tid >> 5;
  const int r0    = lane & 15;
  const int hi4   = (lane >> 4) * 4;
  const int mtile = wv & 3;
  const int ngrp  = wv >> 2;

  v8f accg0 = {}, accg1 = {}, accu0 = {}, accu1 = {};

  load_stage(0);
  store_stage(0);
  const int KT = H_DIM / TILE_K;   // 128
  for (int kt = 0; kt < KT; ++kt) {
    const int buf = kt & 1;
    __syncthreads();
    const bool more = (kt + 1 < KT);
    if (more) load_stage(kt + 1);

    v16bf af   = ld_frag(&As [buf][mtile * 16 + r0][0], hi4);
    v16bf bf00 = ld_frag(&B0s[buf][(ngrp * 2 + 0) * 16 + r0][0], hi4);
    v16bf bf01 = ld_frag(&B0s[buf][(ngrp * 2 + 1) * 16 + r0][0], hi4);
    v16bf bf10 = ld_frag(&B1s[buf][(ngrp * 2 + 0) * 16 + r0][0], hi4);
    v16bf bf11 = ld_frag(&B1s[buf][(ngrp * 2 + 1) * 16 + r0][0], hi4);

    accg0 = __builtin_amdgcn_wmma_f32_16x16x32_bf16(false, af, false, bf00, (short)0, accg0, false, false);
    accg1 = __builtin_amdgcn_wmma_f32_16x16x32_bf16(false, af, false, bf01, (short)0, accg1, false, false);
    accu0 = __builtin_amdgcn_wmma_f32_16x16x32_bf16(false, af, false, bf10, (short)0, accu0, false, false);
    accu1 = __builtin_amdgcn_wmma_f32_16x16x32_bf16(false, af, false, bf11, (short)0, accu1, false, false);

    __syncthreads();
    if (more) store_stage(buf ^ 1);
  }

  // epilogue: silu(g) * u * gate, store bf16
  const int hi8 = (lane >> 4) * 8;
#pragma unroll
  for (int j = 0; j < 2; ++j) {
    v8f g = j ? accg1 : accg0;
    v8f u = j ? accu1 : accu0;
    const int ncol = n0 + (ngrp * 2 + j) * 16 + r0;
#pragma unroll
    for (int r = 0; r < 8; ++r) {
      const int row = mtile * 16 + hi8 + r;
      float gv = g[r], uv = u[r];
      float val = (gv / (1.0f + __expf(-gv))) * uv * sw[row];
      hb[(size_t)(pairbase + row) * I_DIM + ncol] =
          __builtin_bit_cast(unsigned short, (__bf16)val);
    }
  }
}

// ---------------------------------------------------------------------------
// G2: out[t, :] += h[p, :] . wo[e, :, :]^T   (atomic f32 scatter-combine)
//   A tile (h, already bf16) is moved with GLOBAL_LOAD_ASYNC_TO_LDS_B128
//   (ASYNCcnt-tracked, no VGPR round trip) when the builtin is available.
// ---------------------------------------------------------------------------
__global__ __launch_bounds__(256)
void g2_kernel(const unsigned short* __restrict__ hb,
               const float* __restrict__ wo,
               const int* __restrict__ tileoff,
               const int* __restrict__ ptok,
               float* __restrict__ out) {
  __shared__ __align__(16) unsigned As[2][TILE_M][20];
  __shared__ __align__(16) unsigned Bs[2][TILE_N][20];
  __shared__ int stok[TILE_M];

  const int mt    = blockIdx.y;
  const int total = tileoff[E_NUM];
  if (mt >= total) return;
  int e = 0;
#pragma unroll
  for (int i = 1; i < E_NUM; ++i) e += (mt >= tileoff[i]) ? 1 : 0;

  const int tid      = threadIdx.x;
  const int pairbase = mt * TILE_M;
  if (tid < TILE_M) stok[tid] = ptok[pairbase + tid];
  __syncthreads();

  const int n0 = blockIdx.x * TILE_N;                       // H columns
  const float* bp = wo + ((size_t)e * H_DIM + n0) * I_DIM;

  const int lrow = tid >> 2;
  const int lq   = tid & 3;
  const unsigned short* hap = hb + (size_t)(pairbase + lrow) * I_DIM + lq * 8;
  const float* bpp = bp + (size_t)lrow * I_DIM + lq * 8;

  uint4 ra, rb;

  auto load_stage = [&](int kt) {       // B always through VGPRs (needs cvt)
    const int k0 = kt * TILE_K;
#if !HAVE_ASYNC_LDS
    ra = *(const uint4*)(hap + k0);                         // h already bf16
#endif
    rb = ld8_cvt(bpp + k0);
    __builtin_prefetch((const void*)(bpp + k0 + 2 * TILE_K), 0, 3);
  };
  auto store_stage = [&](int kt, int buf) {
#if HAVE_ASYNC_LDS
    __builtin_amdgcn_global_load_async_to_lds_b128(
        (gas_v4i*)(hap + kt * TILE_K),
        (las_v4i*)&As[buf][lrow][lq * 4], 0, 0);
#else
    (void)kt;
    *(uint4*)&As[buf][lrow][lq * 4] = ra;
#endif
    *(uint4*)&Bs[buf][lrow][lq * 4] = rb;
  };

  const int lane  = tid & 31;
  const int wv    = tid >> 5;
  const int r0    = lane & 15;
  const int hi4   = (lane >> 4) * 4;
  const int mtile = wv & 3;
  const int ngrp  = wv >> 2;

  v8f acc0 = {}, acc1 = {};

  load_stage(0);
  store_stage(0, 0);
  const int KT = I_DIM / TILE_K;   // 64
  for (int kt = 0; kt < KT; ++kt) {
    const int buf = kt & 1;
#if HAVE_ASYNC_LDS
    wait_asynccnt0();               // this wave's async tile writes are done
#endif
    __syncthreads();                // => whole 4KB A tile + B tile visible
    const bool more = (kt + 1 < KT);
    if (more) load_stage(kt + 1);

    v16bf af  = ld_frag(&As[buf][mtile * 16 + r0][0], hi4);
    v16bf bf0 = ld_frag(&Bs[buf][(ngrp * 2 + 0) * 16 + r0][0], hi4);
    v16bf bf1 = ld_frag(&Bs[buf][(ngrp * 2 + 1) * 16 + r0][0], hi4);

    acc0 = __builtin_amdgcn_wmma_f32_16x16x32_bf16(false, af, false, bf0, (short)0, acc0, false, false);
    acc1 = __builtin_amdgcn_wmma_f32_16x16x32_bf16(false, af, false, bf1, (short)0, acc1, false, false);

    __syncthreads();
    if (more) store_stage(kt + 1, buf ^ 1);
  }

  const int hi8 = (lane >> 4) * 8;
#pragma unroll
  for (int j = 0; j < 2; ++j) {
    v8f a = j ? acc1 : acc0;
    const int ncol = n0 + (ngrp * 2 + j) * 16 + r0;
#pragma unroll
    for (int r = 0; r < 8; ++r) {
      const int row = mtile * 16 + hi8 + r;
      const int tok = stok[row];
      if (tok >= 0) atomicAdd(&out[(size_t)tok * H_DIM + ncol], a[r]);
    }
  }
}

// ---------------------------------------------------------------------------
// launch: ws layout  [counts 32B | cursor 32B | tileoff 64B | ptok | pw | h(bf16)]
//         total ws need ~= 35.72 MB
// ---------------------------------------------------------------------------
extern "C" void kernel_launch(void* const* d_in, const int* in_sizes, int n_in,
                              void* d_out, int out_size, void* d_ws, size_t ws_size,
                              hipStream_t stream) {
  (void)in_sizes; (void)n_in; (void)ws_size;
  const float* x   = (const float*)d_in[0];
  const float* tw  = (const float*)d_in[1];
  const int*   ti  = (const int*)d_in[2];
  const float* wi0 = (const float*)d_in[3];
  const float* wi1 = (const float*)d_in[4];
  const float* wo  = (const float*)d_in[5];
  float* out = (float*)d_out;

  char* wsb = (char*)d_ws;
  int*   counts  = (int*)(wsb + 0);
  int*   cursor  = (int*)(wsb + 32);
  int*   tileoff = (int*)(wsb + 64);
  int*   ptok    = (int*)(wsb + 128);
  float* pw      = (float*)(wsb + 128 + MAX_PAIRS * 4);
  unsigned short* hb = (unsigned short*)(wsb + 128 + MAX_PAIRS * 8 + 128); // 16B-aligned

  (void)hipMemsetAsync(wsb, 0, 128, stream);                              // counts+cursor+tileoff
  (void)hipMemsetAsync(ptok, 0xFF, MAX_PAIRS * sizeof(int), stream);      // pad slots -> -1
  (void)hipMemsetAsync(out, 0, (size_t)out_size * sizeof(float), stream); // accumulate target

  const int npairs = T_TOK * K_TOP;
  count_kernel<<<(npairs + 255) / 256, 256, 0, stream>>>(ti, counts);
  sched_kernel<<<1, 32, 0, stream>>>(counts, tileoff);
  fill_kernel<<<(npairs + 255) / 256, 256, 0, stream>>>(ti, tw, tileoff, cursor, ptok, pw);

  g1_kernel<<<dim3(I_DIM / TILE_N, MAX_TILES), 256, 0, stream>>>(x, wi0, wi1, tileoff, ptok, pw, hb);
  g2_kernel<<<dim3(H_DIM / TILE_N, MAX_TILES), 256, 0, stream>>>(hb, wo, tileoff, ptok, out);
}
